// HashGridMLPEncoder_30803505447179
// MI455X (gfx1250) — matrix operand order repaired and checked by reading, not compiled, not run on hardware
//
#include <hip/hip_runtime.h>
#include <hip/hip_bf16.h>
#include <stdint.h>

typedef __attribute__((ext_vector_type(16))) _Float16 v16h;
typedef __attribute__((ext_vector_type(8)))  _Float16 v8h;
typedef __attribute__((ext_vector_type(8)))  float    v8f;

#define DIM          3
#define N_LEVELS     16
#define N_FEATS      2
#define HASHMAP_SIZE 32768
#define HIDDEN       64
#define OUT_F        64
#define N_POINTS     524288
#define PTS_PER_BLK  128
#define THREADS      256

// res[i] = int(16 * (2^(1/3))^i)
__device__ __constant__ float kRes[N_LEVELS] = {
    16.f, 20.f, 25.f, 32.f, 40.f, 50.f, 64.f, 80.f,
    101.f, 128.f, 161.f, 203.f, 256.f, 322.f, 406.f, 512.f
};

__device__ inline v16h make_a(v8h lo, v8h hi) {
    v16h r;
#pragma unroll
    for (int i = 0; i < 8; ++i) { r[i] = lo[i]; r[8 + i] = hi[i]; }
    return r;
}

__global__ __launch_bounds__(THREADS, 2)
void hashgrid_mlp_kernel(const float* __restrict__ x,
                         const float* __restrict__ tables,
                         const float* __restrict__ W1,
                         const float* __restrict__ b1,
                         const float* __restrict__ W2,
                         const float* __restrict__ b2,
                         float* __restrict__ out)
{
    // ---- LDS staging ----
    __shared__ float xs[PTS_PER_BLK * DIM];                       // 1.5 KB
    __shared__ __align__(16) _Float16 encS[PTS_PER_BLK][40];      // 32 enc halfs, pad->40
    __shared__ __align__(16) _Float16 w1t[HIDDEN][40];            // W1^T: [n][k], k<32
    __shared__ __align__(16) _Float16 w2t[OUT_F][72];             // W2^T: [n][k], k<64
    __shared__ __align__(16) _Float16 hS[PTS_PER_BLK][72];        // hidden, pad->72
    __shared__ float b1s[HIDDEN];
    __shared__ float b2s[OUT_F];

    const int tid  = threadIdx.x;
    const int lane = tid & 31;
    const int wv   = tid >> 5;                  // 8 waves, each owns 16 points
    const long pblk = (long)blockIdx.x * PTS_PER_BLK;

    // ---- Phase 0: stage x, convert weights to f16 transposed, biases ----
#pragma unroll 2
    for (int i = tid; i < PTS_PER_BLK * DIM; i += THREADS)
        xs[i] = x[pblk * DIM + i];

#pragma unroll 2
    for (int i = tid; i < 32 * HIDDEN; i += THREADS) {
        int k = i >> 6, n = i & 63;
        w1t[n][k] = (_Float16)W1[i];
    }
#pragma unroll 4
    for (int i = tid; i < HIDDEN * OUT_F; i += THREADS) {
        int k = i >> 6, n = i & 63;
        w2t[n][k] = (_Float16)W2[i];
    }
    if (tid < HIDDEN) { b1s[tid] = b1[tid]; b2s[tid] = b2[tid]; }
    __syncthreads();

    // ---- Phase 1: hash-grid encoding -> encS (f16), 2048 tasks / 256 thr ----
#pragma unroll
    for (int t = tid; t < PTS_PER_BLK * N_LEVELS; t += THREADS) {
        const int p   = t >> 4;
        const int lvl = t & 15;
        const float res = kRes[lvl];

        float sx = xs[p * 3 + 0] * res;
        float sy = xs[p * 3 + 1] * res;
        float sz = xs[p * 3 + 2] * res;
        float fx0 = floorf(sx), fy0 = floorf(sy), fz0 = floorf(sz);
        int ix = (int)fx0, iy = (int)fy0, iz = (int)fz0;
        float fx = sx - fx0, fy = sy - fy0, fz = sz - fz0;

        const float2* tab = (const float2*)(tables) + (size_t)lvl * HASHMAP_SIZE;
        float a0 = 0.f, a1 = 0.f;
#pragma unroll
        for (int c = 0; c < 8; ++c) {
            const int cx = c & 1, cy = (c >> 1) & 1, cz = (c >> 2) & 1;
            uint32_t h = (uint32_t)(ix + cx)
                       ^ ((uint32_t)(iy + cy) * 2654435761u)
                       ^ ((uint32_t)(iz + cz) * 805459861u);
            h &= (HASHMAP_SIZE - 1);
            float2 f = tab[h];
            float w = (cx ? fx : 1.f - fx) * (cy ? fy : 1.f - fy) * (cz ? fz : 1.f - fz);
            a0 = fmaf(w, f.x, a0);
            a1 = fmaf(w, f.y, a1);
        }
        encS[p][lvl * 2 + 0] = (_Float16)a0;
        encS[p][lvl * 2 + 1] = (_Float16)a1;
    }
    __syncthreads();

    // ---- Phase 2: layer 1 GEMM via WMMA: [16 x 32] @ [32 x 64] + b1, ReLU ----
    // A fragment (16-bit 16x32): lane<16 -> row M=lane, halfs K{0..7},{16..23}
    //                            lane>=16 -> row M=lane-16, halfs K{8..15},{24..31}
    const int M   = lane & 15;
    const int hi16 = lane >> 4;                // 0 or 1
    const int prow = wv * 16 + M;              // this lane's A row (point)
    const int c0   = hi16 * 8;                 // first K-half offset

    v16h aEnc;
    {
        v8h lo = *(const v8h*)&encS[prow][c0];
        v8h hh = *(const v8h*)&encS[prow][c0 + 16];
        aEnc = make_a(lo, hh);
    }

#pragma unroll
    for (int nt = 0; nt < 4; ++nt) {
        const int ncol = nt * 16 + M;          // lane's output column
        const int k0   = hi16 * 8;             // B: lane<16 K{0..7},{16..23}, lane>=16 K{8..15},{24..31}
        v8h blo = *(const v8h*)&w1t[ncol][k0];
        v8h bhi = *(const v8h*)&w1t[ncol][k0 + 16];
        v16h bFrag = make_a(blo, bhi);

        float bv = b1s[ncol];
        v8f acc;
#pragma unroll
        for (int i = 0; i < 8; ++i) acc[i] = bv;

        acc = __builtin_amdgcn_wmma_f32_16x16x32_f16(
                false, aEnc, false, bFrag, (short)0, acc, false, false);

        // ReLU + f16, scatter into hS: element r -> point (wv*16 + hi16*8 + r), col ncol
        const int prow0 = wv * 16 + hi16 * 8;
#pragma unroll
        for (int r = 0; r < 8; ++r)
            hS[prow0 + r][ncol] = (_Float16)fmaxf(acc[r], 0.f);
    }
    __syncthreads();

    // ---- Phase 3: layer 2 GEMM via WMMA: [16 x 64] @ [64 x 64] + b2 ----
    v16h aH[2];
#pragma unroll
    for (int kc = 0; kc < 2; ++kc) {
        v8h lo = *(const v8h*)&hS[prow][kc * 32 + c0];
        v8h hh = *(const v8h*)&hS[prow][kc * 32 + c0 + 16];
        aH[kc] = make_a(lo, hh);
    }

#pragma unroll
    for (int nt = 0; nt < 4; ++nt) {
        const int ncol = nt * 16 + M;
        float bv = b2s[ncol];
        v8f acc;
#pragma unroll
        for (int i = 0; i < 8; ++i) acc[i] = bv;

#pragma unroll
        for (int kc = 0; kc < 2; ++kc) {
            const int k0 = kc * 32 + hi16 * 8;
            v8h blo = *(const v8h*)&w2t[ncol][k0];
            v8h bhi = *(const v8h*)&w2t[ncol][k0 + 16];
            v16h bFrag = make_a(blo, bhi);
            acc = __builtin_amdgcn_wmma_f32_16x16x32_f16(
                    false, aH[kc], false, bFrag, (short)0, acc, false, false);
        }

        // store: element r -> out[pblk + wv*16 + hi16*8 + r][ncol]
        const long pbase = pblk + wv * 16 + hi16 * 8;
#pragma unroll
        for (int r = 0; r < 8; ++r)
            out[(pbase + r) * OUT_F + ncol] = acc[r];
    }
}

extern "C" void kernel_launch(void* const* d_in, const int* in_sizes, int n_in,
                              void* d_out, int out_size, void* d_ws, size_t ws_size,
                              hipStream_t stream) {
    (void)in_sizes; (void)n_in; (void)out_size; (void)d_ws; (void)ws_size;
    const float* x      = (const float*)d_in[0];
    const float* tables = (const float*)d_in[1];
    const float* W1     = (const float*)d_in[2];
    const float* b1     = (const float*)d_in[3];
    const float* W2     = (const float*)d_in[4];
    const float* b2     = (const float*)d_in[5];
    float* out = (float*)d_out;

    dim3 grid(N_POINTS / PTS_PER_BLK);   // 4096 blocks
    dim3 block(THREADS);                 // 8 waves, 128 points/block
    hashgrid_mlp_kernel<<<grid, block, 0, stream>>>(x, tables, W1, b1, W2, b2, out);
}
